// MSnet_65197603553901
// MI455X (gfx1250) — matrix operand order repaired and checked by reading, not compiled, not run on hardware
//
#include <hip/hip_runtime.h>
#include <math.h>

#define B_    64
#define S_    256
#define L_    4
#define H_    1024
#define SDIM  16
#define FIVE_H (5 * H_)
#define NROWS (B_ * S_ * L_)   // 65536 rows of length H per tensor
#define LN_EPS 1e-5f
#define BN_EPS 1e-5f
#define SM_EPS 1e-15f

typedef __attribute__((ext_vector_type(16))) _Float16 v16h;
typedef __attribute__((ext_vector_type(4)))  _Float16 v4h;
typedef __attribute__((ext_vector_type(2)))  _Float16 v2h;
typedef __attribute__((ext_vector_type(8)))  float    v8f;
typedef __attribute__((ext_vector_type(4)))  float    v4f;

// ---------------------------------------------------------------------------
// Kernel 1: per-(b,s,l) row moments in ONE pass over a/b:
//   sum = Σ_h x,  sq = Σ_h x²,  dot = Σ_h x·p[b,l,h]
// One wave32 per 4 KB row; 8 float4 loads per lane. blockIdx.y selects a vs b.
// ---------------------------------------------------------------------------
__global__ __launch_bounds__(256)
void k1_moments(const float* __restrict__ a, const float* __restrict__ b,
                const float* __restrict__ p,
                float* __restrict__ sumO, float* __restrict__ sqO,
                float* __restrict__ dotO) {
    const int tid  = threadIdx.x;
    const int wave = tid >> 5;
    const int lane = tid & 31;
    const int row  = blockIdx.x * 8 + wave;          // (b*S + s)*L + l
    const int t    = blockIdx.y;                     // 0 = a, 1 = b
    const float* __restrict__ src = t ? b : a;
    const int bb = row / (S_ * L_);
    const int l  = row % L_;
    const float* __restrict__ xv = src + (size_t)row * H_;
    const float* __restrict__ pv = p + ((size_t)bb * L_ + l) * H_;

    float s0 = 0.f, s1 = 0.f, s2 = 0.f;
#pragma unroll
    for (int i = 0; i < 8; ++i) {
        const int h = (lane + i * 32) * 4;
        v4f x = *(const v4f*)(xv + h);
        v4f q = *(const v4f*)(pv + h);
#pragma unroll
        for (int j = 0; j < 4; ++j) {
            s0 += x[j];
            s1 += x[j] * x[j];
            s2 += x[j] * q[j];
        }
    }
    // wave32 butterfly reduce
#pragma unroll
    for (int m = 16; m > 0; m >>= 1) {
        s0 += __shfl_xor(s0, m, 32);
        s1 += __shfl_xor(s1, m, 32);
        s2 += __shfl_xor(s2, m, 32);
    }
    if (lane == 0) {
        const size_t o = (size_t)t * NROWS + row;
        sumO[o] = s0;
        sqO[o]  = s1;
        dotO[o] = s2;
    }
}

// ---------------------------------------------------------------------------
// Kernel 2: per batch b — mask (cumprod of "any nonzero"), masked softmax,
// fused weights  w_s = attn_s*rstd_s  and  c = Σ_s attn_s*m_s*rstd_s.
// One block of 256 threads (= S) per b; handles both tensors, all 4 layers.
// ---------------------------------------------------------------------------
__device__ __forceinline__ float blockReduceSum(float v, float* red) {
    const int tid = threadIdx.x;
    red[tid] = v;
    __syncthreads();
#pragma unroll
    for (int off = 128; off > 0; off >>= 1) {
        if (tid < off) red[tid] += red[tid + off];
        __syncthreads();
    }
    float r = red[0];
    __syncthreads();
    return r;
}

__global__ __launch_bounds__(256)
void k2_softmax(const float* __restrict__ p,
                const float* __restrict__ sumM, const float* __restrict__ sqM,
                const float* __restrict__ dotM,
                float* __restrict__ wOut, float* __restrict__ cOut) {
    __shared__ float red[256];
    __shared__ float s_sump[L_];
    __shared__ int   fz[2];
    const int b = blockIdx.x;
    const int s = threadIdx.x;
    const float invH    = 1.0f / (float)H_;
    const float rescale = 1.0f / 32.0f;              // 1/sqrt(1024)

    // Σ_h p[b,l,h]
    for (int l = 0; l < L_; ++l) {
        const float* pv = p + ((size_t)b * L_ + l) * H_;
        float part = 0.f;
        for (int h = s; h < H_; h += 256) part += pv[h];
        float tot = blockReduceSum(part, red);
        if (s == 0) s_sump[l] = tot;
    }
    if (s < 2) fz[s] = S_;
    __syncthreads();

    // mask: first seq position whose whole [L,H] slice is zero (Σx² == 0)
    for (int t = 0; t < 2; ++t) {
        float tot = 0.f;
        for (int l = 0; l < L_; ++l)
            tot += sqM[(size_t)t * NROWS + ((size_t)b * S_ + s) * L_ + l];
        if (tot == 0.f) atomicMin(&fz[t], s);
    }
    __syncthreads();

    for (int t = 0; t < 2; ++t) {
        const float mask = (s < fz[t]) ? 1.f : 0.f;
        for (int l = 0; l < L_; ++l) {
            const size_t r = (size_t)t * NROWS + ((size_t)b * S_ + s) * L_ + l;
            const float m    = sumM[r] * invH;
            const float var  = sqM[r] * invH - m * m;
            const float rstd = rsqrtf(var + LN_EPS);
            const float sa   = (dotM[r] - m * s_sump[l]) * rstd * rescale;
            const float e    = __expf(sa) * mask;
            const float denom = blockReduceSum(e, red) + SM_EPS;
            const float attn  = e / denom;
            wOut[(((size_t)t * B_ + b) * L_ + l) * S_ + s] = attn * rstd;
            const float ctot = blockReduceSum(attn * m * rstd, red);
            if (s == 0) cOut[((size_t)t * B_ + b) * L_ + l] = ctot;
        }
    }
}

// ---------------------------------------------------------------------------
// Kernel 3: pass 2 — pooled vectors and 5H feature vector (stored as f16).
//   pa[h] = Σ_s w_s · a[b,s,l,h] − c      (ditto pb)
// One block per (b,l); each thread owns 4 consecutive h (float4 / b128 loads,
// 4 KB per tensor per s-step across the block), with global_prefetch to hide
// the 16 KB stride between s-steps. Feature stores are packed 8-byte v4h.
// ---------------------------------------------------------------------------
__global__ __launch_bounds__(256)
void k3_pool_feat(const float* __restrict__ a, const float* __restrict__ b,
                  const float* __restrict__ p,
                  const float* __restrict__ wIn, const float* __restrict__ cIn,
                  _Float16* __restrict__ feat) {
    __shared__ float s_wa[S_], s_wb[S_];
    const int x   = blockIdx.x;                      // b*4 + l
    const int l   = x & 3;
    const int bb  = x >> 2;
    const int tid = threadIdx.x;

    s_wa[tid] = wIn[(((size_t)0 * B_ + bb) * L_ + l) * S_ + tid];
    s_wb[tid] = wIn[(((size_t)1 * B_ + bb) * L_ + l) * S_ + tid];
    __syncthreads();

    const int h = tid * 4;                           // 4 floats per thread
    const size_t stride = (size_t)L_ * H_;           // step between s values
    const size_t base   = ((size_t)bb * S_ * L_ + l) * H_ + h;
    const float* __restrict__ av = a + base;
    const float* __restrict__ bv = b + base;

    v4f acca = {0.f, 0.f, 0.f, 0.f};
    v4f accb = {0.f, 0.f, 0.f, 0.f};
    for (int s = 0; s < S_; ++s) {
        if (s + 8 < S_) {
            __builtin_prefetch(av + (size_t)(s + 8) * stride, 0, 0); // global_prefetch_b8
            __builtin_prefetch(bv + (size_t)(s + 8) * stride, 0, 0);
        }
        const float wa = s_wa[s];
        const float wb = s_wb[s];
        v4f xa = *(const v4f*)(av + (size_t)s * stride);
        v4f xb = *(const v4f*)(bv + (size_t)s * stride);
#pragma unroll
        for (int j = 0; j < 4; ++j) {
            acca[j] += wa * xa[j];
            accb[j] += wb * xb[j];
        }
    }
    const float ca = cIn[((size_t)0 * B_ + bb) * L_ + l];
    const float cb = cIn[((size_t)1 * B_ + bb) * L_ + l];
    v4f pv = *(const v4f*)(p + ((size_t)bb * L_ + l) * H_ + h);

    v4h fp, fa, fb, fpa, fpb;
#pragma unroll
    for (int j = 0; j < 4; ++j) {
        const float pa = acca[j] - ca;
        const float pb = accb[j] - cb;
        fp[j]  = (_Float16)pv[j];
        fa[j]  = (_Float16)pa;
        fb[j]  = (_Float16)pb;
        fpa[j] = (_Float16)(pv[j] * pa);
        fpb[j] = (_Float16)(pv[j] * pb);
    }
    _Float16* f = feat + ((size_t)bb * L_ + l) * FIVE_H;
    *(v4h*)(f + h)          = fp;
    *(v4h*)(f + H_ + h)     = fa;
    *(v4h*)(f + 2 * H_ + h) = fb;
    *(v4h*)(f + 3 * H_ + h) = fpa;
    *(v4h*)(f + 4 * H_ + h) = fpb;
}

// ---------------------------------------------------------------------------
// Kernel 4: per-layer FFNN via v_wmma_f32_16x16x32_f16, fused BN(eval)+ReLU.
// GEMM per layer: [64 x 5120] · [5120 x 16]. Grid = 4 blocks (one per layer),
// 128 threads = 4 waves; wave w owns batch rows [16w, 16w+16). K loop of 160,
// unrolled x2; both WMMAs per iteration chain through the one accumulator.
// Fragment packing follows ISA §7.12.2 (16-bit A 16x32; B 32x16; f32 C 16x16).
// ---------------------------------------------------------------------------
__global__ __launch_bounds__(128)
void k4_ffnn_wmma(const _Float16* __restrict__ feat,
                  const float* __restrict__ ffnn_w, const float* __restrict__ ffnn_b,
                  const float* __restrict__ bn_g, const float* __restrict__ bn_be,
                  const float* __restrict__ bn_rm, const float* __restrict__ bn_rv,
                  float* __restrict__ yBuf) {
    const int l    = blockIdx.x;
    const int wave = threadIdx.x >> 5;               // M-tile 0..3
    const int lane = threadIdx.x & 31;
    const int half = lane >> 4;                      // lane group
    const int mn   = lane & 15;                      // M row (A) / N col (B,C)

    v8f acc = {0.f, 0.f, 0.f, 0.f, 0.f, 0.f, 0.f, 0.f};

    const _Float16* __restrict__ arow =
        feat + (((size_t)(wave * 16 + mn)) * L_ + l) * FIVE_H;
    const float* __restrict__ wmat = ffnn_w + (size_t)l * FIVE_H * SDIM;

#pragma unroll 2
    for (int K0 = 0; K0 < FIVE_H; K0 += 32) {
        v16h A, Bm;
        // A 16x32 f16: lane half 0 -> K pairs {0,2,4,6,16,18,20,22},
        //              lane half 1 -> +8. (ISA 16-bit A layout)
#pragma unroll
        for (int v = 0; v < 8; ++v) {
            const int k = K0 + ((v >> 2) << 4) + (half << 3) + ((v & 3) << 1);
            v2h pr = *(const v2h*)(arow + k);
            A[2 * v]     = pr[0];
            A[2 * v + 1] = pr[1];
        }
        // B 32x16 f16: same K order as A (half 0 -> K {0..7,16..23}, half 1 -> +8)
        const int kb = half * 8;
#pragma unroll
        for (int v = 0; v < 8; ++v) {
            const int k = K0 + kb + ((v >> 2) << 4) + ((v & 3) << 1);
            Bm[2 * v]     = (_Float16)wmat[(size_t)k * SDIM + mn];
            Bm[2 * v + 1] = (_Float16)wmat[(size_t)(k + 1) * SDIM + mn];
        }
        acc = __builtin_amdgcn_wmma_f32_16x16x32_f16(
            /*neg_a=*/false, A, /*neg_b=*/false, Bm,
            /*c_mod=*/(short)0, acc, /*reuse_a=*/false, /*reuse_b=*/false);
    }

    // C layout: VGPR r, lanes 0-15 -> M=r, N=lane; lanes 16-31 -> M=r+8.
    const int   idx   = l * SDIM + mn;               // position in [64]-vector
    const float bias  = ffnn_b[idx];
    const float scale = rsqrtf(bn_rv[idx] + BN_EPS) * bn_g[idx];
    const float mean  = bn_rm[idx];
    const float shift = bn_be[idx];
#pragma unroll
    for (int r = 0; r < 8; ++r) {
        const int m  = r + half * 8;
        const int bb = wave * 16 + m;
        float y = acc[r] + bias;
        y = (y - mean) * scale + shift;
        y = fmaxf(y, 0.f);
        yBuf[(size_t)bb * 64 + idx] = y;
    }
}

// ---------------------------------------------------------------------------
// Kernel 5: distance encodings + final 66 -> 3 head. One thread per batch row.
// ---------------------------------------------------------------------------
__global__ __launch_bounds__(64)
void k5_head(const float* __restrict__ yBuf,
             const float* __restrict__ ap, const float* __restrict__ bp,
             const float* __restrict__ ms_w, const float* __restrict__ ms_b,
             const float* __restrict__ dw, const float* __restrict__ db,
             float* __restrict__ out) {
    const int b = threadIdx.x;
    if (b >= B_) return;
    const float ape = tanhf(ap[b] * dw[0] + db[0]);
    const float bpe = tanhf(bp[b] * dw[0] + db[0]);
    float a0 = ms_b[0], a1 = ms_b[1], a2 = ms_b[2];
    for (int j = 0; j < 64; ++j) {
        const float y = yBuf[(size_t)b * 64 + j];
        a0 += y * ms_w[j * 3 + 0];
        a1 += y * ms_w[j * 3 + 1];
        a2 += y * ms_w[j * 3 + 2];
    }
    a0 += ape * ms_w[64 * 3 + 0] + bpe * ms_w[65 * 3 + 0];
    a1 += ape * ms_w[64 * 3 + 1] + bpe * ms_w[65 * 3 + 1];
    a2 += ape * ms_w[64 * 3 + 2] + bpe * ms_w[65 * 3 + 2];
    out[b * 3 + 0] = a0;
    out[b * 3 + 1] = a1;
    out[b * 3 + 2] = a2;
}

// ---------------------------------------------------------------------------
extern "C" void kernel_launch(void* const* d_in, const int* in_sizes, int n_in,
                              void* d_out, int out_size, void* d_ws, size_t ws_size,
                              hipStream_t stream) {
    const float* a      = (const float*)d_in[0];
    const float* b      = (const float*)d_in[1];
    const float* p      = (const float*)d_in[2];
    const float* ap     = (const float*)d_in[3];
    const float* bp     = (const float*)d_in[4];
    const float* ffnn_w = (const float*)d_in[5];
    const float* ffnn_b = (const float*)d_in[6];
    const float* bn_g   = (const float*)d_in[7];
    const float* bn_be  = (const float*)d_in[8];
    const float* bn_rm  = (const float*)d_in[9];
    const float* bn_rv  = (const float*)d_in[10];
    const float* ms_w   = (const float*)d_in[11];
    const float* ms_b   = (const float*)d_in[12];
    const float* dist_w = (const float*)d_in[13];
    const float* dist_b = (const float*)d_in[14];
    float* out = (float*)d_out;

    // workspace carve (byte offsets; all 1 KB aligned)
    char* ws = (char*)d_ws;
    float*    sumM = (float*)(ws + 0);               // [2][NROWS]
    float*    sqM  = (float*)(ws + 524288);          // [2][NROWS]
    float*    dotM = (float*)(ws + 1048576);         // [2][NROWS]
    float*    wBuf = (float*)(ws + 1572864);         // [2][B][L][S]
    float*    cBuf = (float*)(ws + 2097152);         // [2][B][L]
    float*    yBuf = (float*)(ws + 2099200);         // [B][64]
    _Float16* feat = (_Float16*)(ws + 2115584);      // [B][L][5H] f16

    k1_moments<<<dim3(NROWS / 8, 2), 256, 0, stream>>>(a, b, p, sumM, sqM, dotM);
    k2_softmax<<<B_, 256, 0, stream>>>(p, sumM, sqM, dotM, wBuf, cBuf);
    k3_pool_feat<<<B_ * L_, 256, 0, stream>>>(a, b, p, wBuf, cBuf, feat);
    k4_ffnn_wmma<<<L_, 128, 0, stream>>>(feat, ffnn_w, ffnn_b,
                                         bn_g, bn_be, bn_rm, bn_rv, yBuf);
    k5_head<<<1, 64, 0, stream>>>(yBuf, ap, bp, ms_w, ms_b, dist_w, dist_b, out);
}
